// AlphaWeightedConv2d_65549790872131
// MI455X (gfx1250) — matrix operand order repaired and checked by reading, not compiled, not run on hardware
//
#include <hip/hip_runtime.h>

// CDNA5 / gfx1250: wave32, v_wmma_f32_16x16x32_bf16 implicit-GEMM conv.
// Each block: one (batch, output-row-pair). 7 waves x 16 columns;
// each wave: 4 M-tiles (cout 0..63) x 2 N-tiles (rows h0, h0+1),
// so every A (weight) operand feeds 2 WMMAs.
typedef __attribute__((ext_vector_type(16))) __bf16 v16bf;
typedef __attribute__((ext_vector_type(8)))  __bf16 v8bf;
typedef __attribute__((ext_vector_type(8)))  float  v8f;

constexpr int Bn   = 32;
constexpr int Cin  = 64;
constexpr int Cout = 64;
constexpr int Hs   = 112;
constexpr int Wsz  = 112;
constexpr int Kk   = 3;
constexpr int HW   = Hs * Wsz;            // 12544

constexpr int WROW  = 114;                // staged x row: w = -1..112 (zero halo)
constexpr int NROWS = 4;                  // input rows h0-1 .. h0+2
constexpr int XPOS  = NROWS * WROW;       // 456 spatial positions
constexpr int CP    = 72;                 // channel pitch: 144B = 36 banks ->
                                          // conflict-free, 16B-aligned b128 loads
constexpr int NT    = 224;                // 7 waves

constexpr int XS_BYTES_PAD   = 68096;     // >= XPOS*CP*2 (65664), = 224*19*16
constexpr int WALL_CHUNKS    = 5376;      // 16B chunks, = 224*24 (uniform copy)
constexpr int WALL_BYTES     = WALL_CHUNKS * 16;     // 86016
constexpr int WPACK_ELEMS    = 9 * Cout * CP;        // 41472 used bf16
constexpr int WPACK_ELEM_PAD = WALL_CHUNKS * 8;      // 43008 incl. zero pad
constexpr int GT_BYTES       = 2 * Cout * 4;         // g[] and bias*g[] tables
constexpr size_t LDS_BYTES   = XS_BYTES_PAD + WALL_BYTES + GT_BYTES; // 154624

// ---- Prep: pack W[cout][cin][kh][kw] f32 -> bf16 [tap][cout][CP] (LDS-ready).
__global__ __launch_bounds__(256)
void pack_weights(const float* __restrict__ Wg, __bf16* __restrict__ wpack)
{
    const int i = blockIdx.x * 256 + threadIdx.x;
    if (i >= WPACK_ELEM_PAD) return;
    __bf16 v = (__bf16)0.0f;
    if (i < WPACK_ELEMS) {
        const int ci  = i % CP;
        const int t   = i / CP;
        const int co  = t & (Cout - 1);
        const int tap = t >> 6;
        if (ci < Cin) {
            const int kh = tap / 3, kw = tap - (tap / 3) * 3;
            v = (__bf16)Wg[((co * Cin + ci) * Kk + kh) * Kk + kw];
        }
    }
    wpack[i] = v;
}

__global__ __launch_bounds__(NT, 1)
void conv_gate_wmma(const float* __restrict__ xg, const __bf16* __restrict__ wpack,
                    const float* __restrict__ bias, const float* __restrict__ alpha,
                    const int* __restrict__ label, float* __restrict__ out)
{
    extern __shared__ char smem[];
    __bf16* xs   = (__bf16*)smem;                    // [pos][CP] x tile (bf16)
    __bf16* wall = (__bf16*)(smem + XS_BYTES_PAD);   // [tap][cout][CP] weights
    float*  gtab = (float*)(smem + XS_BYTES_PAD + WALL_BYTES);        // g[co]
    float*  btab = gtab + Cout;                                       // bias*g

    const int tid    = threadIdx.x;
    const int wave   = tid >> 5;
    const int lane   = tid & 31;
    const int lanelo = lane & 15;
    const int hi     = lane >> 4;

    const int b  = blockIdx.x / (Hs / 2);
    const int h0 = (blockIdx.x - b * (Hs / 2)) * 2;  // output rows h0, h0+1
    const int w  = wave * 16 + lanelo;

    // ---- Gate tables: g = sigmoid(alpha[label[b], co]), bg = bias*g.
    if (tid < Cout) {
        const float a = alpha[(size_t)label[b] * Cout + tid];
        const float g = 1.0f / (1.0f + __expf(-a));
        gtab[tid] = g;
        btab[tid] = bias[tid] * g;
    }
    // ---- Zero-fill x tile (halo + OOB rows): b128 DS stores, 19/thread.
    {
        uint4* xz = (uint4*)xs;
        const uint4 z = {0u, 0u, 0u, 0u};
#pragma unroll
        for (int i = 0; i < XS_BYTES_PAD / 16 / NT; ++i) xz[i * NT + tid] = z;
    }
    // ---- Stage all 9 taps of prepacked weights: b128 copy, 24/thread.
    {
        const uint4* src = (const uint4*)wpack;
        uint4*       dst = (uint4*)wall;
#pragma unroll 4
        for (int i = tid; i < WALL_CHUNKS; i += NT) dst[i] = src[i];
    }
    __syncthreads();

    // ---- Stage x interior: float4 loads (coalesced, aligned), bf16 scatter.
    //      NROWS*64*28 = 7168 groups = 224*32 (uniform). OOB rows: clamp+mask.
    {
        const float4* xb4 = (const float4*)(xg + (size_t)b * Cin * HW);
#pragma unroll 4
        for (int g = tid; g < NROWS * Cin * 28; g += NT) {
            const int rr  = g / (Cin * 28);
            const int rem = g - rr * (Cin * 28);
            const int c   = rem / 28;
            const int q   = rem - c * 28;            // float4 index within row
            const int hr  = h0 - 1 + rr;
            const int hrc = min(max(hr, 0), Hs - 1);
            const float m = (hr == hrc) ? 1.0f : 0.0f;
            const float4 v = xb4[c * (HW / 4) + hrc * (Wsz / 4) + q];
            const int base = (rr * WROW + q * 4 + 1) * CP + c;
            xs[base         ] = (__bf16)(v.x * m);
            xs[base +     CP] = (__bf16)(v.y * m);
            xs[base + 2 * CP] = (__bf16)(v.z * m);
            xs[base + 3 * CP] = (__bf16)(v.w * m);
        }
    }
    __syncthreads();

    v8f acc0[4] = {};       // row h0   : 4 M-tiles of 16x16 f32
    v8f acc1[4] = {};       // row h0+1

#pragma unroll 3
    for (int p = 0; p < 9; ++p) {           // 3x3 taps, no barriers
        const int kh = p / 3;
        const int kw = p - kh * 3;
        const __bf16* xrow = &xs[(kh * WROW + (w + kw)) * CP];
        const __bf16* wtap = &wall[p * Cout * CP];

#pragma unroll
        for (int c0 = 0; c0 < Cin; c0 += 32) {      // two K=32 steps per tap
            // B operands 32x16 for both output rows:
            // half lo = K c0..c0+15, half hi = K c0+16..c0+31.
            const __bf16* bp = xrow + c0 + 16 * hi;
            const v8bf b0lo = *(const v8bf*)(bp);
            const v8bf b0hi = *(const v8bf*)(bp + 8);
            const v8bf b1lo = *(const v8bf*)(bp + WROW * CP);
            const v8bf b1hi = *(const v8bf*)(bp + WROW * CP + 8);
            v16bf bv0, bv1;
#pragma unroll
            for (int e = 0; e < 8; ++e) {
                bv0[e] = b0lo[e]; bv0[e + 8] = b0hi[e];
                bv1[e] = b1lo[e]; bv1[e + 8] = b1hi[e];
            }

            v16bf av[4];
#pragma unroll
            for (int mt = 0; mt < 4; ++mt) {
                const __bf16* row = &wtap[(mt * 16 + lanelo) * CP + c0 + 16 * hi];
                const v8bf a_lo = *(const v8bf*)(row);       // K = c0+16hi  ..+7
                const v8bf a_hi = *(const v8bf*)(row + 8);   // K = c0+16hi+8..+7
#pragma unroll
                for (int e = 0; e < 8; ++e) { av[mt][e] = a_lo[e]; av[mt][e + 8] = a_hi[e]; }
            }
#pragma unroll
            for (int mt = 0; mt < 4; ++mt) {    // each A feeds 2 WMMAs
                acc0[mt] = __builtin_amdgcn_wmma_f32_16x16x32_bf16(
                    false, av[mt], false, bv0, (short)0, acc0[mt], false, false);
                acc1[mt] = __builtin_amdgcn_wmma_f32_16x16x32_bf16(
                    false, av[mt], false, bv1, (short)0, acc1[mt], false, false);
            }
        }
    }

    // ---- Epilogue: y = acc*g + bias*g via LDS tables.
    // C/D layout: VGPR j, lane -> M = j + 8*hi, N = lanelo.
    float* outr = out + ((size_t)b * Cout) * HW + h0 * Wsz + w;
#pragma unroll
    for (int mt = 0; mt < 4; ++mt) {
#pragma unroll
        for (int j = 0; j < 8; ++j) {
            const int co = mt * 16 + j + 8 * hi;
            const float g  = gtab[co];
            const float bg = btab[co];
            outr[(size_t)co * HW      ] = acc0[mt][j] * g + bg;
            outr[(size_t)co * HW + Wsz] = acc1[mt][j] * g + bg;
        }
    }
}

extern "C" void kernel_launch(void* const* d_in, const int* in_sizes, int n_in,
                              void* d_out, int out_size, void* d_ws, size_t ws_size,
                              hipStream_t stream) {
    (void)in_sizes; (void)n_in; (void)out_size; (void)ws_size;
    const float* x     = (const float*)d_in[0];
    const float* W     = (const float*)d_in[1];
    const float* bias  = (const float*)d_in[2];
    const float* alpha = (const float*)d_in[3];
    const int*   label = (const int*)d_in[4];
    float*  outp  = (float*)d_out;
    __bf16* wpack = (__bf16*)d_ws;                 // 86016 B of workspace

    hipLaunchKernelGGL(pack_weights, dim3((WPACK_ELEM_PAD + 255) / 256), dim3(256),
                       0, stream, W, wpack);
    hipLaunchKernelGGL(conv_gate_wmma, dim3(Bn * (Hs / 2)), dim3(NT),
                       LDS_BYTES, stream, x, wpack, bias, alpha, label, outp);
}
